// AttModel_61984968015884
// MI455X (gfx1250) — compile-verified, matrix operand order checked by reading, not run on hardware
//
#include <hip/hip_runtime.h>
#include <hip/hip_bf16.h>
#include <stdint.h>

// ---------------------------------------------------------------------------
// Problem constants (from reference): B=8, S=96, D=768, R=24, RSEQ=8, TAG=3
// ---------------------------------------------------------------------------
#define BSZ   8
#define SEQ   96
#define DIM   768
#define RNUM  24
#define RSEQL 8
#define TAGSZ 3
#define N3D   2304          // 3*D  (K of the big GEMM)
#define NOUT  72            // R*TAG
#define NPAD  80            // 72 padded to 5*16
#define N2    4608          // 2*N3D (Hh|Ht concatenated)

typedef __attribute__((ext_vector_type(16))) __bf16 v16bf;
typedef __attribute__((ext_vector_type(8)))  float  v8f;

union FragU { uint4 u[2]; v16bf v; };

// f32 -> bf16, round-to-nearest-even
__device__ __forceinline__ uint16_t f2bf(float f) {
  uint32_t u = __float_as_uint(f);
  uint32_t r = (u + 0x7FFFu + ((u >> 16) & 1u)) >> 16;
  return (uint16_t)r;
}

// CDNA5 async global->LDS copy (ASYNCcnt-tracked, bypasses VGPRs).
// lds_off: wave-relative LDS byte offset (low 32 bits of a generic pointer
// derived from an AS(3) object); g: per-lane global address.
__device__ __forceinline__ void async_ld_b128(uint32_t lds_off, const void* g) {
  asm volatile("global_load_async_to_lds_b128 %0, %1, off"
               :: "v"(lds_off), "v"(g) : "memory");
}
__device__ __forceinline__ void wait_async0() {
  asm volatile("s_wait_asynccnt 0x0" ::: "memory");
}

// Load one 16x32 bf16 WMMA fragment (A-layout; B uses the same pattern when
// the matrix is stored N-major).  Per CDNA5 ISA 7.12.2:
//   lanes 0-15 : row = lane,    VGPR0-3 = K[0..8),  VGPR4-7 = K[16..24)
//   lanes 16-31: row = lane-16, VGPR0-3 = K[8..16), VGPR4-7 = K[24..32)
__device__ __forceinline__ v16bf frag_ld(const uint16_t* rowbase, int ldk, int lane) {
  const int r  = lane & 15;
  const int hi = (lane >> 4) & 1;
  const uint16_t* p = rowbase + r * ldk + 8 * hi;
  FragU t;
  t.u[0] = *(const uint4*)(p);        // K chunk 0 (8 bf16 = 16B)
  t.u[1] = *(const uint4*)(p + 16);   // K chunk 1
  return t.v;
}

// ---------------------------------------------------------------------------
// Stage 1: iterative refinement.  One workgroup per sample; whole 96x768
// state + 8x768 A live in the 320KB WGP LDS (row stride 772 avoids the
// 768 % 64banks == 0 conflict pathology).  f32 math: errors compound over 24
// sequential steps, so no low precision here.
// ---------------------------------------------------------------------------
#define AST 772
__global__ __launch_bounds__(768) void k_refine(const float* __restrict__ enc,
                                                const float* __restrict__ rel,
                                                uint16_t* __restrict__ enc_bf) {
  extern __shared__ float sm[];
  float* sA = sm;                 // 8  * AST
  float* sB = sm + 8 * AST;       // 96 * AST
  float* sS = sm + 104 * AST;     // 8 * 96 scores / softmax weights
  const int b = blockIdx.x, tid = threadIdx.x;

  for (int r = 0; r < RSEQL; ++r) sA[r * AST + tid] = rel[(b * RSEQL + r) * DIM + tid];
  for (int s = 0; s < SEQ;   ++s) sB[s * AST + tid] = enc[(b * SEQ  + s) * DIM + tid];
  __syncthreads();

  const float scale = 1.0f / sqrtf((float)DIM);
  for (int step = 0; step < RNUM; ++step) {
    // scores = A @ b^T * scale   (768 threads == 8*96 dot products)
    {
      const int r = tid / SEQ, s = tid % SEQ;
      const float* ar = sA + r * AST;
      const float* br = sB + s * AST;
      float acc = 0.f;
      for (int k = 0; k < DIM; ++k) acc = fmaf(ar[k], br[k], acc);
      sS[r * SEQ + s] = acc * scale;
    }
    __syncthreads();
    // softmax over s (rows are tiny: 8 rows x 96)
    if (tid < RSEQL) {
      float* row = sS + tid * SEQ;
      float m = -3.4e38f;
      for (int s = 0; s < SEQ; ++s) m = fmaxf(m, row[s]);
      float sum = 0.f;
      for (int s = 0; s < SEQ; ++s) { float e = expf(row[s] - m); row[s] = e; sum += e; }
      float inv = 1.0f / sum;
      for (int s = 0; s < SEQ; ++s) row[s] *= inv;
    }
    __syncthreads();
    // b += w^T @ A   (per thread: fixed k = tid, sweep s)
    for (int s = 0; s < SEQ; ++s) {
      float acc = 0.f;
      #pragma unroll
      for (int r = 0; r < RSEQL; ++r) acc = fmaf(sS[r * SEQ + s], sA[r * AST + tid], acc);
      sB[s * AST + tid] += acc;
    }
    __syncthreads();
  }
  for (int s = 0; s < SEQ; ++s)
    enc_bf[(b * SEQ + s) * DIM + tid] = f2bf(sB[s * AST + tid]);
}

// ---------------------------------------------------------------------------
// Prep: proj_W (1536 x 2304, k-major) -> N-major bf16 [4608][768] so that the
// WMMA B fragment load uses the A-fragment lane pattern.
// ---------------------------------------------------------------------------
__global__ void k_prep_projw(const float* __restrict__ W, uint16_t* __restrict__ Wt) {
  const int idx = blockIdx.x * 256 + threadIdx.x;       // n*768 + k, exact grid
  const int n = idx / DIM, k = idx % DIM;
  const int half = (n >= N3D) ? 1 : 0;
  Wt[idx] = f2bf(W[(half * DIM + k) * N3D + (n - half * N3D)]);
}

// rel_W (2304 x 72) -> N-major bf16 [80][2304], cols 72..79 zero-padded.
__global__ void k_prep_relw(const float* __restrict__ W, uint16_t* __restrict__ Wt) {
  const int idx = blockIdx.x * 256 + threadIdx.x;       // n*2304 + k, exact grid
  const int n = idx / N3D, k = idx % N3D;
  Wt[idx] = (n < NOUT) ? f2bf(W[k * NOUT + n]) : (uint16_t)0;
}

// ---------------------------------------------------------------------------
// Stage 2: H[768][4608] = enc_bf16 @ [Wh|Wt].  One 16x16 tile per wave,
// K=768 -> 24 v_wmma_f32_16x16x32_bf16.  Sources are ~8MB -> L2 resident;
// the compiler pipelines global_load_b128 clauses against the WMMAs.
// ---------------------------------------------------------------------------
__global__ __launch_bounds__(256) void k_gemm2(const uint16_t* __restrict__ A,
                                               const uint16_t* __restrict__ Bm,
                                               float* __restrict__ H) {
  const int lane = threadIdx.x & 31, wave = threadIdx.x >> 5;
  const int tile = blockIdx.x * 8 + wave;               // 48*288 tiles, exact
  const int mt = tile / (N2 / 16), nt = tile % (N2 / 16);
  const uint16_t* arow = A  + (mt * 16) * DIM;
  const uint16_t* brow = Bm + (nt * 16) * DIM;
  v8f acc = {};
  for (int k0 = 0; k0 < DIM; k0 += 32) {
    v16bf a = frag_ld(arow + k0, DIM, lane);
    v16bf b = frag_ld(brow + k0, DIM, lane);
    acc = __builtin_amdgcn_wmma_f32_16x16x32_bf16(false, a, false, b,
                                                  (short)0, acc, false, false);
  }
  // C/D layout: VGPR v -> M = v + 8*(lane>=16), N = lane%16
  const int hi = lane >> 4, n = nt * 16 + (lane & 15);
  #pragma unroll
  for (int v = 0; v < 8; ++v)
    H[(mt * 16 + v + 8 * hi) * N2 + n] = acc[v];
}

// ---------------------------------------------------------------------------
// Stage 3 (the big one, fused): out = transpose(relu(Hh_i + Ht_j + pb) @ relW
// + rb).  The 680MB `pairs` tensor is never materialized.  Each WG owns one
// (b, 16 i's, 16 j's) block = 256 rows x 80 cols.  Per K=32 chunk the Hh/Ht
// slices and the relW^T chunk for chunk it+1 are fetched with ASYNC
// global->LDS B128 copies (double-buffered), overlapping the DMA with the
// VALU relu-pair build and the 8x10 WMMAs of chunk it.
// ---------------------------------------------------------------------------
#define ASTRIDE 40   // 32 bf16 + 8 pad -> 80B rows, 16B aligned, low conflicts
#define HST     36   // 32 f32 + 4 pad  -> 144B rows, 16B aligned, no conflicts
__global__ __launch_bounds__(256) void k_pairs(const float* __restrict__ H,
                                               const float* __restrict__ pb,
                                               const uint16_t* __restrict__ Wt,
                                               const float* __restrict__ rb,
                                               float* __restrict__ out) {
  __shared__ __align__(16) float    sHh[2][16][HST];
  __shared__ __align__(16) float    sHt[2][16][HST];
  __shared__ __align__(16) float    sPb[N3D];
  __shared__ __align__(16) uint16_t sA[256 * ASTRIDE];    // 256 rows x 32 bf16
  __shared__ __align__(16) uint16_t sB[2][NPAD][ASTRIDE]; // 80 n-rows x 32 bf16

  const int i0 = blockIdx.x * 16, j0 = blockIdx.y * 16, b = blockIdx.z;
  const int tid = threadIdx.x, lane = tid & 31, wave = tid >> 5;
  const int il = tid >> 4, jl = tid & 15;

  const float* Hh = H + (b * SEQ + i0) * N2;            // cols [0, 2304)
  const float* Ht = H + (b * SEQ + j0) * N2 + N3D;      // cols [2304, 4608)

  // Issue one chunk's worth of async global->LDS fills into buffer `bufi`.
  auto stage = [&](int bufi, int k0) {
    // Hh/Ht: 16 rows x 32 f32 = 128 B128 transfers each (one per thread)
    {
      const int r = (tid & 127) >> 3, c = tid & 7;
      const float* g = ((tid < 128) ? Hh : Ht) + r * N2 + k0 + c * 4;
      float* l = (tid < 128) ? &sHh[bufi][r][c * 4] : &sHt[bufi][r][c * 4];
      async_ld_b128((uint32_t)(uintptr_t)l, g);
    }
    // relW^T chunk: 80 rows x 32 bf16 = 320 B128 transfers
    {
      int n = tid >> 2, c = tid & 3;
      async_ld_b128((uint32_t)(uintptr_t)&sB[bufi][n][c * 8],
                    Wt + n * N3D + k0 + c * 8);
      if (tid < 64) {
        const int idx = tid + 256;
        n = idx >> 2; c = idx & 3;
        async_ld_b128((uint32_t)(uintptr_t)&sB[bufi][n][c * 8],
                      Wt + n * N3D + k0 + c * 8);
      }
    }
  };

  v8f acc[2][5] = {};

  // prologue: chunk 0 fills + bias preload (whole 2304-vector, once)
  stage(0, 0);
  #pragma unroll
  for (int idx = tid; idx < N3D; idx += 256) sPb[idx] = pb[idx];
  wait_async0();
  __syncthreads();

  for (int it = 0; it < N3D / 32; ++it) {
    const int cur = it & 1;
    const int k0 = it * 32;
    if (it + 1 < N3D / 32) stage(cur ^ 1, k0 + 32);   // overlap next chunk DMA

    // build A row (il, jl): relu(Hh[i] + Ht[j] + pb) -> bf16, packed stores
    {
      uint32_t* arow = (uint32_t*)(sA + (il * 16 + jl) * ASTRIDE);
      const float* hh  = &sHh[cur][il][0];
      const float* ht  = &sHt[cur][jl][0];
      const float* pbk = &sPb[k0];
      #pragma unroll
      for (int kk = 0; kk < 32; kk += 2) {
        float f0 = fmaxf(hh[kk]     + ht[kk]     + pbk[kk],     0.f);
        float f1 = fmaxf(hh[kk + 1] + ht[kk + 1] + pbk[kk + 1], 0.f);
        arow[kk >> 1] = (uint32_t)f2bf(f0) | ((uint32_t)f2bf(f1) << 16);
      }
    }
    __syncthreads();
    // compute: each wave -> 2 i-tiles x 5 n-tiles
    v16bf bfr[5];
    #pragma unroll
    for (int nt = 0; nt < 5; ++nt)
      bfr[nt] = frag_ld(&sB[cur][nt * 16][0], ASTRIDE, lane);
    #pragma unroll
    for (int t = 0; t < 2; ++t) {
      v16bf af = frag_ld(sA + ((wave * 2 + t) * 16) * ASTRIDE, ASTRIDE, lane);
      #pragma unroll
      for (int nt = 0; nt < 5; ++nt)
        acc[t][nt] = __builtin_amdgcn_wmma_f32_16x16x32_bf16(
            false, af, false, bfr[nt], (short)0, acc[t][nt], false, false);
    }
    if (it + 1 < N3D / 32) wait_async0();  // this wave's next-chunk fills done
    __syncthreads();                       // all waves done reading cur bufs/sA
  }

  // epilogue: out[b, t3, r, i, j] = acc + rel_b[n];  n = r*TAG + t3
  const int hi = lane >> 4, nl = lane & 15;
  #pragma unroll
  for (int nt = 0; nt < 5; ++nt) {
    const int n = nt * 16 + nl;
    if (n >= NOUT) continue;                  // padded columns
    const float rbn = rb[n];
    const int r = n / TAGSZ, t3 = n % TAGSZ;
    #pragma unroll
    for (int t = 0; t < 2; ++t) {
      const int i = i0 + wave * 2 + t;        // M-tile index == local i
      float* orow = out + ((((b * TAGSZ + t3) * RNUM + r) * SEQ + i) * SEQ) + j0;
      #pragma unroll
      for (int v = 0; v < 8; ++v)             // D rows == local j
        orow[v + 8 * hi] = acc[t][nt][v] + rbn;
    }
  }
}

// ---------------------------------------------------------------------------
// Launch
// ---------------------------------------------------------------------------
extern "C" void kernel_launch(void* const* d_in, const int* in_sizes, int n_in,
                              void* d_out, int out_size, void* d_ws, size_t ws_size,
                              hipStream_t stream) {
  (void)in_sizes; (void)n_in; (void)out_size; (void)ws_size;
  const float* enc = (const float*)d_in[0];   // (8, 96, 768)
  const float* rel = (const float*)d_in[1];   // (24, 8, 768)
  const float* pW  = (const float*)d_in[2];   // (1536, 2304)
  const float* pb  = (const float*)d_in[3];   // (2304,)
  const float* rW  = (const float*)d_in[4];   // (2304, 72)
  const float* rb  = (const float*)d_in[5];   // (72,)
  float* out = (float*)d_out;                 // (8, 3, 24, 96, 96)

  // workspace layout (all 16B aligned); total ~22.8 MB
  char* ws = (char*)d_ws;
  float*    Hbuf   = (float*)(ws);                     // 768*4608 f32 = 14,155,776
  uint16_t* enc_bf = (uint16_t*)(ws + 14155776);       // 768*768 bf16 =  1,179,648
  uint16_t* projWt = (uint16_t*)(ws + 15335424);       // 4608*768 bf16 = 7,077,888
  uint16_t* relWt  = (uint16_t*)(ws + 22413312);       // 80*2304 bf16 =    368,640

  const size_t lds_refine = (size_t)(104 * AST + 8 * 96) * sizeof(float); // ~317KB of 320KB WGP LDS
  k_refine<<<8, 768, lds_refine, stream>>>(enc, rel, enc_bf);
  k_prep_projw<<<(N2 * DIM) / 256, 256, 0, stream>>>(pW, projWt);
  k_prep_relw<<<(NPAD * N3D) / 256, 256, 0, stream>>>(rW, relWt);
  k_gemm2<<<(48 * 288) / 8, 256, 0, stream>>>(enc_bf, projWt, Hbuf);
  k_pairs<<<dim3(SEQ / 16, SEQ / 16, BSZ), 256, 0, stream>>>(Hbuf, pb, relWt, rb, out);
}
